// SimpleGCN_4844723109935
// MI455X (gfx1250) — matrix operand order repaired and checked by reading, not compiled, not run on hardware
//
#include <hip/hip_runtime.h>
#include <hip/hip_bf16.h>

// ---------------------------------------------------------------------------
// GCN pipeline for MI455X (gfx1250, wave32, WMMA).
// Dense GEMMs -> v_wmma_f32_16x16x32_bf16 with vectorized LDS fragment loads;
// edge scatter -> native fp32 atomics (unsafeAtomicAdd).
// ---------------------------------------------------------------------------

typedef __attribute__((ext_vector_type(16))) __bf16 v16bf;
typedef __attribute__((ext_vector_type(8)))  __bf16 v8bf;
typedef __attribute__((ext_vector_type(8)))  float  v8f;

union FragAB { v16bf v; v8bf half8[2]; __bf16 e[16]; };
union FragC  { v8f   v; float  e[8];  };

__device__ __forceinline__ __bf16 f2bf(float f) { return (__bf16)f; }

__device__ __forceinline__ void atomAddF(float* p, float v) {
    unsafeAtomicAdd(p, v);   // native global_atomic_add_f32
}

#define GEMM_N 64

// C[M x 64] = bf16(maybe_relu(X[M x K])) @ bf16(W[K x 64]); fp32 accumulate.
// Block = 256 threads = 8 waves; block tile = 32 rows x 64 cols;
// wave tile = 16x16; K stepped by 32 per WMMA.
//
// sX:  row-major [32 x K]; lane's A fragment = two contiguous 16B chunks.
// sWf: fragment-major: [(kt*4+nt)*32 + lane]*16 + i  -> each lane's B
//      fragment is 32 contiguous bytes (two ds_load_b128).
__global__ __launch_bounds__(256) void gcn_gemm_wmma(
    const float* __restrict__ X, const float* __restrict__ Wg,
    float* __restrict__ Out, int M, int K, int relu_in)
{
    __shared__ __align__(32) __bf16 sWf[128 * GEMM_N];  // up to K=128
    __shared__ __align__(32) __bf16 sX[32 * 128];

    const int tid = threadIdx.x;
    const int r0  = blockIdx.x * 32;

    // Stage W into LDS, permuted to B-fragment-major order.
    // Global element W[k][ncol]: kt=k/32, w=k%32, h=(w/8)%2, i=w%8+8*(w/16),
    // nt=ncol/16, n=ncol%16, lane=h*16+n  ->  sWf[((kt*4+nt)*32 + lane)*16 + i].
    for (int idx = tid; idx < K * GEMM_N; idx += 256) {
        int k = idx >> 6, ncol = idx & 63;
        int kt = k >> 5, w = k & 31;
        int h = (w >> 3) & 1, i = (w & 7) + ((w >> 4) << 3);
        int nt = ncol >> 4, n = ncol & 15;
        int lane = (h << 4) + n;
        sWf[((((kt << 2) + nt) << 5) + lane) * 16 + i] = f2bf(Wg[idx]);
    }

    // Stage 32 rows of X; zero-pad rows past M so WMMA runs with EXEC all-1s.
    for (int idx = tid; idx < 32 * K; idx += 256) {
        int row = idx / K, col = idx - row * K;
        int gr  = r0 + row;
        float v = (gr < M) ? X[(size_t)gr * K + col] : 0.0f;
        if (relu_in) v = fmaxf(v, 0.0f);
        sX[idx] = f2bf(v);
    }
    __syncthreads();

    const int wave = tid >> 5;
    const int lane = tid & 31;
    const int mt = wave >> 2;          // 0..1  (M sub-tile)
    const int nt = wave & 3;           // 0..3  (N sub-tile)
    const int h  = lane >> 4;          // lane half
    const int n  = lane & 15;          // column within tile (B/C/D)
    const int m  = lane & 15;          // row within tile (A)

    FragC acc;
#pragma unroll
    for (int i = 0; i < 8; ++i) acc.e[i] = 0.0f;

    for (int k0 = 0; k0 < K; k0 += 32) {
        FragAB a, b;
        // A fragment (16-bit A-matrix 16x32 layout, ISA 7.12.2):
        // lane holds K = k0+h*8 .. +7 and k0+16+h*8 .. +7 for row m.
        const __bf16* abase = &sX[(mt * 16 + m) * K + k0 + (h << 3)];
        a.half8[0] = *(const v8bf*)(abase);
        a.half8[1] = *(const v8bf*)(abase + 16);

        // B fragment: 32 contiguous bytes per lane in fragment-major sWf.
        const int kt = k0 >> 5;
        b.v = *(const v16bf*)&sWf[((((kt << 2) + nt) << 5) + lane) * 16];

        acc.v = __builtin_amdgcn_wmma_f32_16x16x32_bf16(
            false, a.v, false, b.v, (short)0, acc.v, false, false);
    }

    // C/D layout: VGPR r -> M = r + 8*h, N = n.
#pragma unroll
    for (int r = 0; r < 8; ++r) {
        int row = r0 + mt * 16 + r + (h << 3);
        if (row < M) Out[(size_t)row * GEMM_N + nt * 16 + n] = acc.e[r];
    }
}

// ---------------------------------------------------------------------------

__global__ void k_fill(float* __restrict__ p, float v, int n) {
    int i = blockIdx.x * blockDim.x + threadIdx.x;
    if (i < n) p[i] = v;
}

__global__ void k_deg_edges(const int* __restrict__ dst,
                            const float* __restrict__ ew,
                            float* __restrict__ deg, int E) {
    int i = blockIdx.x * blockDim.x + threadIdx.x;
    if (i < E) atomAddF(&deg[dst[i]], ew[i]);
}

__global__ void k_dinv(float* __restrict__ deg, int n) {
    int i = blockIdx.x * blockDim.x + threadIdx.x;
    if (i < n) {
        float d = deg[i];
        deg[i] = (d > 0.0f) ? rsqrtf(d) : 0.0f;
    }
}

__global__ void k_norm(const int* __restrict__ src, const int* __restrict__ dst,
                       const float* __restrict__ ew, const float* __restrict__ dinv,
                       float* __restrict__ nrm, int E) {
    int i = blockIdx.x * blockDim.x + threadIdx.x;
    if (i < E) nrm[i] = dinv[src[i]] * ew[i] * dinv[dst[i]];
}

// acc[i,c] = xw[i,c] * dinv[i]^2 (self-loop) + bias[c]
__global__ void k_selfinit(const float* __restrict__ xw, const float* __restrict__ dinv,
                           const float* __restrict__ bias, float* __restrict__ out, int N) {
    int i = blockIdx.x * blockDim.x + threadIdx.x;
    if (i < N * GEMM_N) {
        int node = i >> 6, c = i & 63;
        float di = dinv[node];
        out[i] = xw[i] * di * di + bias[c];
    }
}

// One thread per (edge, channel-pair): gather float2, two fp32 atomics.
__global__ void k_scatter(const int* __restrict__ src, const int* __restrict__ dst,
                          const float* __restrict__ nrm, const float* __restrict__ xw,
                          float* __restrict__ acc, int E) {
    int t = blockIdx.x * blockDim.x + threadIdx.x;
    if (t < E * 32) {
        int e = t >> 5;
        int c = (t & 31) << 1;
        int s = src[e], d = dst[e];
        float nv = nrm[e];
        const float2 xv = *(const float2*)(xw + (size_t)s * GEMM_N + c);
        atomAddF(&acc[(size_t)d * GEMM_N + c],     nv * xv.x);
        atomAddF(&acc[(size_t)d * GEMM_N + c + 1], nv * xv.y);
    }
}

__global__ void k_pool(const int* __restrict__ batch, const float* __restrict__ acc,
                       float* __restrict__ gsum, float* __restrict__ gcnt, int N) {
    int t = blockIdx.x * blockDim.x + threadIdx.x;
    if (t < N * GEMM_N) {
        int node = t >> 6, c = t & 63;
        int g = batch[node];
        atomAddF(&gsum[g * GEMM_N + c], acc[t]);
        if (c == 0) atomAddF(&gcnt[g], 1.0f);
    }
}

// One block (64 threads) per graph: pooled -> 32 -> 10.
__global__ __launch_bounds__(64) void k_head(
    const float* __restrict__ gsum, const float* __restrict__ gcnt,
    const float* __restrict__ LW1, const float* __restrict__ Lb1,
    const float* __restrict__ LW2, const float* __restrict__ Lb2,
    float* __restrict__ out) {
    int g = blockIdx.x;
    int t = threadIdx.x;
    __shared__ float p[64];
    __shared__ float h1[32];
    float inv = 1.0f / fmaxf(gcnt[g], 1.0f);
    p[t] = gsum[g * 64 + t] * inv;
    __syncthreads();
    if (t < 32) {
        float s = Lb1[t];
        for (int k = 0; k < 64; ++k) s += p[k] * LW1[k * 32 + t];
        h1[t] = s;
    }
    __syncthreads();
    if (t < 10) {
        float s = Lb2[t];
        for (int k = 0; k < 32; ++k) s += h1[k] * LW2[k * 10 + t];
        out[g * 10 + t] = s;
    }
}

// ---------------------------------------------------------------------------

extern "C" void kernel_launch(void* const* d_in, const int* in_sizes, int n_in,
                              void* d_out, int out_size, void* d_ws, size_t ws_size,
                              hipStream_t stream) {
    const int N = 50000, E = 800000, G = 512;
    const int IN_CH = 128;

    const float* x     = (const float*)d_in[0];
    const int*   ei    = (const int*)  d_in[1];   // [2, E]
    const float* ew    = (const float*)d_in[2];
    const int*   batch = (const int*)  d_in[3];
    const float* W1    = (const float*)d_in[4];
    const float* b1    = (const float*)d_in[5];
    const float* W2    = (const float*)d_in[6];
    const float* b2    = (const float*)d_in[7];
    const float* LW1   = (const float*)d_in[8];
    const float* Lb1   = (const float*)d_in[9];
    const float* LW2   = (const float*)d_in[10];
    const float* Lb2   = (const float*)d_in[11];
    float* out = (float*)d_out;

    const int* src = ei;
    const int* dst = ei + E;

    // Workspace layout (floats, 256B aligned chunks).
    auto alignup = [](size_t v) { return (v + 63) & ~(size_t)63; };
    float* p     = (float*)d_ws;
    float* dinv  = p; p += alignup((size_t)N);
    float* nrm   = p; p += alignup((size_t)E);
    float* xw    = p; p += alignup((size_t)N * GEMM_N);   // reused: xw1 then xw2
    float* acc   = p; p += alignup((size_t)N * GEMM_N);   // reused: acc1 then acc2
    float* gsum  = p; p += alignup((size_t)G * GEMM_N);
    float* gcnt  = p; p += alignup((size_t)G);
    (void)ws_size; (void)in_sizes; (void)n_in; (void)out_size;

    const int B = 256;
    auto blks = [](long long n, int b) { return (unsigned)((n + b - 1) / b); };

    // Degree (self-loop weight 1) -> dinv.
    k_fill<<<blks(N, B), B, 0, stream>>>(dinv, 1.0f, N);
    k_deg_edges<<<blks(E, B), B, 0, stream>>>(dst, ew, dinv, E);
    k_dinv<<<blks(N, B), B, 0, stream>>>(dinv, N);
    k_norm<<<blks(E, B), B, 0, stream>>>(src, dst, ew, dinv, nrm, E);

    const unsigned gemmGrid = blks(N, 32);

    // ---- Layer 1: xw1 = x @ W1 ; acc1 = scatter + selfloop + b1 ----
    gcn_gemm_wmma<<<gemmGrid, B, 0, stream>>>(x, W1, xw, N, IN_CH, 0);
    k_selfinit<<<blks((long long)N * GEMM_N, B), B, 0, stream>>>(xw, dinv, b1, acc, N);
    k_scatter<<<blks((long long)E * 32, B), B, 0, stream>>>(src, dst, nrm, xw, acc, E);

    // ---- Layer 2: xw2 = relu(acc1) @ W2 (relu fused into load) ----
    gcn_gemm_wmma<<<gemmGrid, B, 0, stream>>>(acc, W2, xw, N, GEMM_N, 1);
    k_selfinit<<<blks((long long)N * GEMM_N, B), B, 0, stream>>>(xw, dinv, b2, acc, N);
    k_scatter<<<blks((long long)E * 32, B), B, 0, stream>>>(src, dst, nrm, xw, acc, E);

    // ---- Mean pool + head ----
    k_fill<<<blks(G * GEMM_N, B), B, 0, stream>>>(gsum, 0.0f, G * GEMM_N);
    k_fill<<<blks(G, B), B, 0, stream>>>(gcnt, 0.0f, G);
    k_pool<<<blks((long long)N * GEMM_N, B), B, 0, stream>>>(batch, acc, gsum, gcnt, N);
    k_head<<<G, 64, 0, stream>>>(gsum, gcnt, LW1, Lb1, LW2, Lb2, out);
}
